// OffsetMSE_72421738545288
// MI455X (gfx1250) — matrix operand order repaired and checked by reading, not compiled, not run on hardware
//
#include <hip/hip_runtime.h>
#include <hip/hip_bf16.h>

typedef __attribute__((ext_vector_type(16))) __bf16 v16bf;
typedef __attribute__((ext_vector_type(8)))  float  v8f;
typedef __attribute__((ext_vector_type(8)))  unsigned int v8u;

#define LFULL   1048576
#define BATCH   16
#define NLAG    96          // lags 0..95 tracked; lag 0 computed but unused
#define TCHUNK  4096        // time samples per block
#define QPAD    240         // q history: all 16 WMMA N-columns index in-bounds
#define PPAD    16          // p tail: A reads up to t0+14 past chunk end
#define NWAVES  8
#define THREADS (NWAVES * 32)
#define KSTEP   32          // K per bf16 WMMA
#define CHUNKS_PER_B (LFULL / TCHUNK)

// bf16 RNE + hi/lo split, packed as u32 {lo[31:16], hi[15:0]}
__device__ __forceinline__ unsigned int bf16_rne(float x) {
    unsigned int u = __float_as_uint(x);
    return (u + 0x7FFFu + ((u >> 16) & 1u)) >> 16;
}
__device__ __forceinline__ unsigned int pack_hilo(float x) {
    unsigned int hi = bf16_rne(x);
    float hif = __uint_as_float(hi << 16);
    unsigned int lo = bf16_rne(x - hif);
    return (lo << 16) | hi;
}

// ---------------------------------------------------------------- zero ws ---
__global__ __launch_bounds__(128) void zero_ws_kernel(float* ws) {
    int t = threadIdx.x;
    if (t < NLAG + 2) ws[t] = 0.0f;
}

// ------------------------------------------------------- main WMMA kernel ---
// ws[0..95] = C(s) summed over batch; ws[96] = sum p^2; ws[97] = sum q^2
__global__ __launch_bounds__(THREADS) void corr_wmma_kernel(
        const float* __restrict__ p, const float* __restrict__ q,
        float* __restrict__ ws) {
    __shared__ __align__(16) unsigned int p32_s[TCHUNK + PPAD];
    __shared__ __align__(16) unsigned int q32_s[TCHUNK + QPAD];
    __shared__ float blockCorr[NLAG];
    __shared__ float blockSP, blockSQ;

    const int tid = threadIdx.x;
    const int b   = blockIdx.x / CHUNKS_PER_B;
    const int c   = blockIdx.x % CHUNKS_PER_B;
    const long cs = (long)c * TCHUNK;
    const float* pb = p + (size_t)b * LFULL;
    const float* qb = q + (size_t)b * LFULL;

    if (tid < NLAG) blockCorr[tid] = 0.0f;
    if (tid == 0) { blockSP = 0.0f; blockSQ = 0.0f; }

    float psum = 0.0f, qsum = 0.0f;

    // Stage p: element j = p[cs + j], zero past L; convert once, reuse ~32x.
    for (int j4 = tid; j4 < (TCHUNK + PPAD) / 4; j4 += THREADS) {
        long g = cs + (long)j4 * 4;
        float4 v = make_float4(0.0f, 0.0f, 0.0f, 0.0f);
        if (g + 3 < LFULL) v = *(const float4*)(pb + g);
        uint4 w;
        w.x = pack_hilo(v.x); w.y = pack_hilo(v.y);
        w.z = pack_hilo(v.z); w.w = pack_hilo(v.w);
        ((uint4*)p32_s)[j4] = w;
        if (j4 < TCHUNK / 4)
            psum += v.x*v.x + v.y*v.y + v.z*v.z + v.w*v.w;
    }
    // Stage q with 240-element history: element j = q[cs - 240 + j], zero below 0.
    for (int j4 = tid; j4 < (TCHUNK + QPAD) / 4; j4 += THREADS) {
        long g = cs - QPAD + (long)j4 * 4;
        float4 v = make_float4(0.0f, 0.0f, 0.0f, 0.0f);
        if (g >= 0) v = *(const float4*)(qb + g);
        uint4 w;
        w.x = pack_hilo(v.x); w.y = pack_hilo(v.y);
        w.z = pack_hilo(v.z); w.w = pack_hilo(v.w);
        ((uint4*)q32_s)[j4] = w;
        if (j4 >= QPAD / 4)
            qsum += v.x*v.x + v.y*v.y + v.z*v.z + v.w*v.w;
    }
    __syncthreads();

    // D[m,n] += sum_k p[cs+t0+k+m] * q[cs+t0+k-16n]   (k = 0..31 per group)
    // bf16 A 16x32: lane m=l%16, h=l/16; VGPR v:   K=2v+16h(+1)
    // bf16 B 32x16: lane n=l%16, h=l/16; VGPR v:   K=2v+16h(+1)
    const int wave = tid / 32;
    const int lane = tid % 32;
    const int mn   = lane & 15;
    const int h    = lane >> 4;
    const int aoff = mn + 16 * h;                       // + t0 -> A run base
    const int boff = QPAD + 16 * h - 16 * mn;           // + t0 -> B run base (>=0)

    v8f acc = {};
    #pragma unroll 2
    for (int i = 0; i < TCHUNK / (KSTEP * NWAVES); ++i) {
        const int t0 = KSTEP * (wave + NWAVES * i);

        // ---- A operands: two 8-word runs ----
        unsigned int w0[8], w1[8];
        #pragma unroll
        for (int j = 0; j < 8; ++j) {
            w0[j] = p32_s[t0 + aoff + j];
            w1[j] = p32_s[t0 + aoff + 8 + j];
        }
        v8u ahu, alu;
        #pragma unroll
        for (int v = 0; v < 4; ++v) {
            ahu[v]     = __builtin_amdgcn_perm(w0[2*v+1], w0[2*v], 0x05040100u);
            alu[v]     = __builtin_amdgcn_perm(w0[2*v+1], w0[2*v], 0x07060302u);
            ahu[v + 4] = __builtin_amdgcn_perm(w1[2*v+1], w1[2*v], 0x05040100u);
            alu[v + 4] = __builtin_amdgcn_perm(w1[2*v+1], w1[2*v], 0x07060302u);
        }
        // ---- B operand: one 16-word run, 64B-aligned -> ds_load_b128 x4 ----
        const unsigned int* qr = q32_s + (t0 + boff);
        uint4 qv0 = ((const uint4*)qr)[0];
        uint4 qv1 = ((const uint4*)qr)[1];
        uint4 qv2 = ((const uint4*)qr)[2];
        uint4 qv3 = ((const uint4*)qr)[3];
        unsigned int wb[16] = { qv0.x, qv0.y, qv0.z, qv0.w,
                                qv1.x, qv1.y, qv1.z, qv1.w,
                                qv2.x, qv2.y, qv2.z, qv2.w,
                                qv3.x, qv3.y, qv3.z, qv3.w };
        v8u bhu, blu;
        #pragma unroll
        for (int v = 0; v < 8; ++v) {
            bhu[v] = __builtin_amdgcn_perm(wb[2*v+1], wb[2*v], 0x05040100u);
            blu[v] = __builtin_amdgcn_perm(wb[2*v+1], wb[2*v], 0x07060302u);
        }

        v16bf ah = __builtin_bit_cast(v16bf, ahu);
        v16bf al = __builtin_bit_cast(v16bf, alu);
        v16bf bh = __builtin_bit_cast(v16bf, bhu);
        v16bf bl = __builtin_bit_cast(v16bf, blu);

        // split-precision fp32 emulation: hi*hi + hi*lo + lo*hi (lo*lo ~2^-18, dropped)
        acc = __builtin_amdgcn_wmma_f32_16x16x32_bf16(false, ah, false, bh,
                                                      (short)0, acc, false, false);
        acc = __builtin_amdgcn_wmma_f32_16x16x32_bf16(false, ah, false, bl,
                                                      (short)0, acc, false, false);
        acc = __builtin_amdgcn_wmma_f32_16x16x32_bf16(false, al, false, bh,
                                                      (short)0, acc, false, false);
    }
    __syncthreads();

    // C/D layout: lane l, VGPR r holds D[M = r + 8*(l/16), N = l%16] -> lag 16*mn + r + 8h
    if (mn < 6) {
        #pragma unroll
        for (int r = 0; r < 8; ++r) {
            int s = 16 * mn + r + 8 * h;
            atomicAdd(&blockCorr[s], acc[r]);
        }
    }
    atomicAdd(&blockSP, psum);
    atomicAdd(&blockSQ, qsum);
    __syncthreads();

    if (tid < NLAG) atomicAdd(&ws[tid], blockCorr[tid]);
    if (tid == 0) {
        atomicAdd(&ws[NLAG],     blockSP);
        atomicAdd(&ws[NLAG + 1], blockSQ);
    }
}

// --------------------------------------------------------- finalize (1 CTA) -
__global__ __launch_bounds__(128) void finalize_kernel(
        const float* __restrict__ p, const float* __restrict__ q,
        const float* __restrict__ ws, float* __restrict__ out) {
    __shared__ float EP[NLAG], EQ[NLAG], LOSS[NLAG];
    const int t = threadIdx.x;

    if (t < NLAG) {
        float ep = 0.0f, eq = 0.0f;
        for (int b = 0; b < BATCH; ++b) {
            float pv = p[(size_t)b * LFULL + t];
            float qv = q[(size_t)b * LFULL + (LFULL - 1 - t)];
            ep += pv * pv;
            eq += qv * qv;
        }
        EP[t] = ep;
        EQ[t] = eq;
    }
    __syncthreads();

    if (t >= 1 && t < NLAG) {
        float pref = 0.0f, suf = 0.0f;
        for (int u = 0; u < t; ++u) { pref += EP[u]; suf += EQ[u]; }
        const float sp = ws[NLAG], sq = ws[NLAG + 1], cc = ws[t];
        const float num = (sp - pref) + (sq - suf) - 2.0f * cc;
        LOSS[t] = num / (16.0f * (float)(LFULL - t));
    }
    __syncthreads();

    if (t == 0) {
        float m = LOSS[1];
        for (int s = 2; s < NLAG; ++s) m = fminf(m, LOSS[s]);
        out[0] = m;
    }
}

// ------------------------------------------------------------------ launch --
extern "C" void kernel_launch(void* const* d_in, const int* in_sizes, int n_in,
                              void* d_out, int out_size, void* d_ws, size_t ws_size,
                              hipStream_t stream) {
    const float* p  = (const float*)d_in[0];  // predict, (16,1,2^20) f32
    const float* q  = (const float*)d_in[1];  // target,  (16,1,2^20) f32
    float* ws  = (float*)d_ws;
    float* out = (float*)d_out;

    zero_ws_kernel<<<1, 128, 0, stream>>>(ws);
    corr_wmma_kernel<<<BATCH * CHUNKS_PER_B, THREADS, 0, stream>>>(p, q, ws);
    finalize_kernel<<<1, 128, 0, stream>>>(p, q, ws, out);
}
